// QMARLNet_70583492543103
// MI455X (gfx1250) — matrix operand order repaired and checked by reading, not compiled, not run on hardware
//
#include <hip/hip_runtime.h>
#include <hip/hip_bf16.h>

// ---------------------------------------------------------------------------
// QMARLNet for MI455X (gfx1250, wave32, WMMA).
//   k_zero      : zero global ensemble accumulators in ws
//   k_enc_nodes : node0[B,N,H]  = relu(W2 relu(W1 LN(obs)))   (dedup over roots)
//   k_enc_edges : edge0[B,N,N,E]= relu(enc_ew ef + b)         (dedup over roots)
//   k_gnn_root  : one workgroup per (b,r); both layers in LDS; WMMA f16 GEMMs;
//                 wave-per-Ntile ownership (no LDS atomics), hoisted B
//                 fragments, ds_load_b128 A fragments, WMMA heads
//   k_final     : counts, eps-mix, renorm, write (probs, ens_q, ens_raw)
// ---------------------------------------------------------------------------

#define BB   8
#define NN   32
#define OBS  64
#define HH   128
#define EE   16
#define LL   2
#define AA   10
#define W2HE (2*HH+EE)     // 272
#define NSTR 132           // padded LDS row stride for fp32 [32][128] tiles

typedef __attribute__((ext_vector_type(16))) _Float16 v16h;
typedef __attribute__((ext_vector_type(8)))  _Float16 v8h;
typedef __attribute__((ext_vector_type(4)))  _Float16 v4h;
typedef __attribute__((ext_vector_type(8)))  float    v8f;

__device__ __forceinline__ v16h mkA(v8h lo, v8h hi) {
    v16h a;
#pragma unroll
    for (int t = 0; t < 8; ++t) { a[t] = lo[t]; a[t + 8] = hi[t]; }
    return a;
}

// Build B fragment from 16 consecutive fp32 weights (K = laneHi*16 + t).
__device__ __forceinline__ v16h mkB16(const float* wp) {
    float4 w0 = *(const float4*)(wp + 0);
    float4 w1 = *(const float4*)(wp + 4);
    float4 w2 = *(const float4*)(wp + 8);
    float4 w3 = *(const float4*)(wp + 12);
    v16h b;
    b[0]=(_Float16)w0.x; b[1]=(_Float16)w0.y; b[2]=(_Float16)w0.z; b[3]=(_Float16)w0.w;
    b[4]=(_Float16)w1.x; b[5]=(_Float16)w1.y; b[6]=(_Float16)w1.z; b[7]=(_Float16)w1.w;
    b[8]=(_Float16)w2.x; b[9]=(_Float16)w2.y; b[10]=(_Float16)w2.z; b[11]=(_Float16)w2.w;
    b[12]=(_Float16)w3.x; b[13]=(_Float16)w3.y; b[14]=(_Float16)w3.z; b[15]=(_Float16)w3.w;
    return b;
}

// ---------------------------------------------------------------- zero accum
__global__ void k_zero(float* p, int n) {
    int i = blockIdx.x * blockDim.x + threadIdx.x;
    if (i < n) p[i] = 0.f;
}

// --------------------------------------------------------------- obs encoder
__global__ __launch_bounds__(128)
void k_enc_nodes(const float* __restrict__ obs,
                 const float* __restrict__ ln_g, const float* __restrict__ ln_b,
                 const float* __restrict__ w1, const float* __restrict__ b1,
                 const float* __restrict__ w2, const float* __restrict__ b2,
                 float* __restrict__ node0) {
    __shared__ float sObs[OBS];
    __shared__ float sLn[OBS];
    __shared__ float sX1[HH];
    __shared__ float sMu, sRstd;
    const int row = blockIdx.x;                // b*N + n
    const int tid = threadIdx.x;
    if (tid < OBS) sObs[tid] = obs[row * OBS + tid];
    __syncthreads();
    if (tid == 0) {
        float mu = 0.f;
        for (int j = 0; j < OBS; ++j) mu += sObs[j];
        mu *= (1.f / OBS);
        float var = 0.f;
        for (int j = 0; j < OBS; ++j) { float d = sObs[j] - mu; var += d * d; }
        var *= (1.f / OBS);
        sMu = mu; sRstd = rsqrtf(var + 1e-5f);
    }
    __syncthreads();
    if (tid < OBS) sLn[tid] = (sObs[tid] - sMu) * sRstd * ln_g[tid] + ln_b[tid];
    __syncthreads();
    {
        float acc = b1[tid];
        const float* wr = w1 + tid * OBS;
        for (int j = 0; j < OBS; ++j) acc += wr[j] * sLn[j];
        sX1[tid] = fmaxf(acc, 0.f);
    }
    __syncthreads();
    {
        float acc = b2[tid];
        const float* wr = w2 + tid * HH;
        for (int j = 0; j < HH; ++j) acc += wr[j] * sX1[j];
        node0[row * HH + tid] = fmaxf(acc, 0.f);
    }
}

// -------------------------------------------------------------- edge encoder
__global__ __launch_bounds__(256)
void k_enc_edges(const float* __restrict__ ef,
                 const float* __restrict__ ew, const float* __restrict__ eb,
                 float* __restrict__ edge0) {
    __shared__ float sEf[NN * EE];
    __shared__ float sW[EE * EE];
    __shared__ float sB[EE];
    const int blk = blockIdx.x;      // b*N + i
    const int tid = threadIdx.x;
    const long base = (long)blk * NN * EE;
    sEf[tid]       = ef[base + tid];
    sEf[tid + 256] = ef[base + tid + 256];
    if (tid < EE * EE) sW[tid] = ew[tid];
    if (tid < EE) sB[tid] = eb[tid];
    __syncthreads();
    for (int out = tid; out < NN * EE; out += 256) {
        int j = out >> 4, e = out & 15;
        float acc = sB[e];
        const float* wr = sW + e * EE;
        const float* xr = sEf + j * EE;
        for (int k = 0; k < EE; ++k) acc += wr[k] * xr[k];
        edge0[base + out] = fmaxf(acc, 0.f);
    }
}

// ------------------------------------------------------- per-root GNN kernel
__global__ __launch_bounds__(256)
void k_gnn_root(const float* __restrict__ adjg,   // [B,N,N]
                const float* __restrict__ smg,    // [B,N,N]
                const float* __restrict__ nr_w, const float* __restrict__ nr_b,
                const float* __restrict__ er_w, const float* __restrict__ er_b,
                const float* __restrict__ no_w, const float* __restrict__ no_b,
                const float* __restrict__ eo_w, const float* __restrict__ eo_b,
                const float* __restrict__ pol_w, const float* __restrict__ pol_b,
                const float* __restrict__ q_w,  const float* __restrict__ q_b,
                const float* __restrict__ node0, const float* __restrict__ edge0,
                float* __restrict__ accR, float* __restrict__ accQ) {
    __shared__ __align__(16) float    sNode [NN * NSTR];       // node fp32
    __shared__ __align__(16) float    sAgg  [NN * NSTR];       // agg fp32
    __shared__ __align__(16) _Float16 sNodeH[NN * HH];         // node f16 shadow
    __shared__ __align__(16) _Float16 sAggH [NN * HH];         // hid  f16 shadow
    __shared__ __align__(16) _Float16 sEdge[2][NN * NN * EE];  // edge ping-pong
    __shared__ __align__(16) float    sAdj [NN * NN];
    __shared__ __align__(16) float    sSrc [NN * EE];          // also head logits
    __shared__ __align__(16) float    sDst [NN * EE];          // also head q
    __shared__ float    sNm  [NN];

    const int b = blockIdx.x >> 5;
    const int r = blockIdx.x & 31;
    const int tid  = threadIdx.x;
    const int wave = tid >> 5;          // wave owns Ntile == wave
    const int lane = tid & 31;
    const int laneLo = lane & 15, laneHi = lane >> 4;
    const v8h zH = {};

    // ---- init: masks -> adjacency -> (node, edge) staging, b128 loads ----
    if (tid < NN) sNm[tid] = smg[(b * NN + r) * NN + tid];
    __syncthreads();
    for (int idx = tid; idx < NN * NN; idx += 256) {
        int i = idx >> 5, j = idx & 31;
        sAdj[idx] = adjg[b * NN * NN + idx] * sNm[i] * sNm[j];
    }
    __syncthreads();                               // sAdj ready for edge staging
    {
        const float4* np4 = (const float4*)(node0 + (long)(b * NN) * HH);
        for (int q = tid; q < NN * HH / 4; q += 256) {   // 4 iters/thread
            int i = q >> 5, c = q & 31;
            float nmv = sNm[i];
            float4 x = np4[q];
            x.x *= nmv; x.y *= nmv; x.z *= nmv; x.w *= nmv;
            *(float4*)&sNode[i * NSTR + c * 4] = x;
            v4h h4; h4[0]=(_Float16)x.x; h4[1]=(_Float16)x.y;
                    h4[2]=(_Float16)x.z; h4[3]=(_Float16)x.w;
            *(v4h*)&sNodeH[i * HH + c * 4] = h4;
        }
        const float4* eg4 = (const float4*)(edge0 + (long)b * NN * NN * EE);
        for (int q = tid; q < NN * NN * EE / 4; q += 256) {  // 16 iters/thread
            float am = sAdj[q >> 2];                 // 4 e-values share one pair
            float4 x = eg4[q];
            v4h h4; h4[0]=(_Float16)(x.x*am); h4[1]=(_Float16)(x.y*am);
                    h4[2]=(_Float16)(x.z*am); h4[3]=(_Float16)(x.w*am);
            *(v4h*)&sEdge[0][q * 4] = h4;
        }
    }
    __syncthreads();

    int cur = 0;
    for (int l = 0; l < LL; ++l) {
        const int h = wave * 16 + laneLo;          // this wave's output column

        // ---- (b) agg[i,h] = sum_j relu(edge[i,j,:] @ er_w[h,:] + er_b) * adj
        {
            v16h bbf;
#pragma unroll
            for (int t = 0; t < 16; ++t) bbf[t] = (_Float16)0.f;
            if (laneHi == 0)
                bbf = mkB16(er_w + ((long)l * HH + h) * EE);   // K=0..15 real
            const float bias = er_b[l * HH + h];
            for (int i = 0; i < NN; ++i) {
                float acc = 0.f;
#pragma unroll
                for (int sub = 0; sub < 2; ++sub) {
                    // A rows: pair = i*32 + sub*16 + laneLo, K(e) contiguous
                    const _Float16* ep =
                        &sEdge[cur][(i * NN + sub * 16 + laneLo) * EE];
                    v16h a = laneHi ? mkA(zH, zH) : mkA(*(const v8h*)ep, *(const v8h*)(ep + 8));
                    v8f c = {};
                    c = __builtin_amdgcn_wmma_f32_16x16x32_f16(false, a, false, bbf,
                                                               (short)0, c, false, false);
                    const float4 am0 = *(const float4*)&sAdj[i * NN + sub * 16 + 8 * laneHi];
                    const float4 am1 = *(const float4*)&sAdj[i * NN + sub * 16 + 8 * laneHi + 4];
                    acc += fmaxf(c[0] + bias, 0.f) * am0.x;
                    acc += fmaxf(c[1] + bias, 0.f) * am0.y;
                    acc += fmaxf(c[2] + bias, 0.f) * am0.z;
                    acc += fmaxf(c[3] + bias, 0.f) * am0.w;
                    acc += fmaxf(c[4] + bias, 0.f) * am1.x;
                    acc += fmaxf(c[5] + bias, 0.f) * am1.y;
                    acc += fmaxf(c[6] + bias, 0.f) * am1.z;
                    acc += fmaxf(c[7] + bias, 0.f) * am1.w;
                }
                float other = __shfl_xor(acc, 16, 32);   // combine lane-half partials
                if (laneHi == 0) sAgg[i * NSTR + h] = acc + other;
            }
        }
        __syncthreads();

        // ---- (c) hid = relu(node @ nr_w^T + nr_b) + agg  -> sAggH (f16)
        {
            const float* wr = nr_w + ((long)l * HH + h) * HH;
            const float bias = nr_b[l * HH + h];
            v8f c0 = {}, c1 = {};
#pragma unroll
            for (int kst = 0; kst < 4; ++kst) {
                const int k0 = kst * 32;
                v16h bbf = mkB16(wr + k0 + laneHi * 16);
#pragma unroll
                for (int m = 0; m < 2; ++m) {
                    const _Float16* np = &sNodeH[(m * 16 + laneLo) * HH + k0 + laneHi * 16];
                    v16h a = mkA(*(const v8h*)np, *(const v8h*)(np + 8));
                    if (m == 0)
                        c0 = __builtin_amdgcn_wmma_f32_16x16x32_f16(false, a, false, bbf,
                                                                    (short)0, c0, false, false);
                    else
                        c1 = __builtin_amdgcn_wmma_f32_16x16x32_f16(false, a, false, bbf,
                                                                    (short)0, c1, false, false);
                }
            }
#pragma unroll
            for (int v = 0; v < 8; ++v) {
                int r0 = v + 8 * laneHi, r1 = 16 + v + 8 * laneHi;
                sAggH[r0 * HH + h] = (_Float16)(fmaxf(c0[v] + bias, 0.f) + sAgg[r0 * NSTR + h]);
                sAggH[r1 * HH + h] = (_Float16)(fmaxf(c1[v] + bias, 0.f) + sAgg[r1 * NSTR + h]);
            }
        }
        __syncthreads();

        // ---- (d) node = (hid @ no_w^T + no_b + node) * nm
        {
            const float* wr = no_w + ((long)l * HH + h) * HH;
            const float bias = no_b[l * HH + h];
            v8f c0 = {}, c1 = {};
#pragma unroll
            for (int kst = 0; kst < 4; ++kst) {
                const int k0 = kst * 32;
                v16h bbf = mkB16(wr + k0 + laneHi * 16);
#pragma unroll
                for (int m = 0; m < 2; ++m) {
                    const _Float16* hp = &sAggH[(m * 16 + laneLo) * HH + k0 + laneHi * 16];
                    v16h a = mkA(*(const v8h*)hp, *(const v8h*)(hp + 8));
                    if (m == 0)
                        c0 = __builtin_amdgcn_wmma_f32_16x16x32_f16(false, a, false, bbf,
                                                                    (short)0, c0, false, false);
                    else
                        c1 = __builtin_amdgcn_wmma_f32_16x16x32_f16(false, a, false, bbf,
                                                                    (short)0, c1, false, false);
                }
            }
#pragma unroll
            for (int v = 0; v < 8; ++v) {
                int r0 = v + 8 * laneHi, r1 = 16 + v + 8 * laneHi;
                float n0 = (c0[v] + bias + sNode[r0 * NSTR + h]) * sNm[r0];
                float n1 = (c1[v] + bias + sNode[r1 * NSTR + h]) * sNm[r1];
                sNode[r0 * NSTR + h] = n0;  sNodeH[r0 * HH + h] = (_Float16)n0;
                sNode[r1 * NSTR + h] = n1;  sNodeH[r1 * HH + h] = (_Float16)n1;
            }
        }
        __syncthreads();

        if (l + 1 < LL) {
            // ---- (e) src_t = node @ ws^T ; dst_t = node @ wd^T  (E=16 cols)
            if (wave < 4) {
                const int which = wave >> 1, mtile = wave & 1;
                const int e = laneLo;
                const float* wr = eo_w + ((long)l * EE + e) * W2HE + (which ? HH : 0);
                v8f c = {};
#pragma unroll
                for (int kst = 0; kst < 4; ++kst) {
                    const int k0 = kst * 32;
                    v16h bbf = mkB16(wr + k0 + laneHi * 16);
                    const _Float16* np =
                        &sNodeH[(mtile * 16 + laneLo) * HH + k0 + laneHi * 16];
                    v16h a = mkA(*(const v8h*)np, *(const v8h*)(np + 8));
                    c = __builtin_amdgcn_wmma_f32_16x16x32_f16(false, a, false, bbf,
                                                               (short)0, c, false, false);
                }
                float* dst = which ? sDst : sSrc;
#pragma unroll
                for (int v = 0; v < 8; ++v)
                    dst[(mtile * 16 + v + 8 * laneHi) * EE + e] = c[v];
            }
            __syncthreads();

            // ---- (f) edge = relu(src[i] + dst[j] + edge @ we^T + eo_b) * adj
            const int nxt = cur ^ 1;
            {
                const int e = laneLo;
                v16h bbf;
#pragma unroll
                for (int t = 0; t < 16; ++t) bbf[t] = (_Float16)0.f;
                if (laneHi == 0)
                    bbf = mkB16(eo_w + ((long)l * EE + e) * W2HE + 2 * HH);
                const float bias = eo_b[l * EE + e];
                for (int mtile = wave; mtile < 64; mtile += 8) {
                    const int row = mtile * 16 + laneLo;
                    const _Float16* ep = &sEdge[cur][row * EE];
                    v16h a = laneHi ? mkA(zH, zH) : mkA(*(const v8h*)ep, *(const v8h*)(ep + 8));
                    v8f c = {};
                    c = __builtin_amdgcn_wmma_f32_16x16x32_f16(false, a, false, bbf,
                                                               (short)0, c, false, false);
#pragma unroll
                    for (int v = 0; v < 8; ++v) {
                        int rr = mtile * 16 + v + 8 * laneHi;
                        int i2 = rr >> 5, j2 = rr & 31;
                        float val = c[v] + sSrc[i2 * EE + e] + sDst[j2 * EE + e] + bias;
                        sEdge[nxt][rr * EE + e] =
                            (_Float16)(fmaxf(val, 0.f) * sAdj[rr]);
                    }
                }
            }
            __syncthreads();
            cur = nxt;
        }
    }

    // ---- heads via WMMA: logits/q = node @ {pol_w,q_w}^T  (A padded 10->16)
    if (wave < 4) {
        const int which = wave >> 1, mtile = wave & 1;   // which: 0=policy 1=q
        const float* hw = which ? q_w : pol_w;
        v16h zb = {};
        v8f c = {};
#pragma unroll
        for (int kst = 0; kst < 4; ++kst) {
            const int k0 = kst * 32;
            v16h bbf = zb;
            if (laneLo < AA)
                bbf = mkB16(hw + laneLo * HH + k0 + laneHi * 16);
            const _Float16* np = &sNodeH[(mtile * 16 + laneLo) * HH + k0 + laneHi * 16];
            v16h a = mkA(*(const v8h*)np, *(const v8h*)(np + 8));
            c = __builtin_amdgcn_wmma_f32_16x16x32_f16(false, a, false, bbf,
                                                       (short)0, c, false, false);
        }
        float* dst = which ? sDst : sSrc;
#pragma unroll
        for (int v = 0; v < 8; ++v)
            dst[(mtile * 16 + v + 8 * laneHi) * EE + laneLo] = c[v];
    }
    __syncthreads();

    // ---- softmax + ensemble scatter (A=10, tiny -> VALU) ----
    if (tid < NN) {
        const int i = tid;
        const float nmv = sNm[i];
        if (nmv != 0.f) {
            float lg[AA], qv[AA];
            for (int a = 0; a < AA; ++a) {
                lg[a] = sSrc[i * EE + a] + pol_b[a];
                qv[a] = sDst[i * EE + a] + q_b[a];
            }
            float mx = lg[0];
            for (int a = 1; a < AA; ++a) mx = fmaxf(mx, lg[a]);
            float s = 0.f;
            for (int a = 0; a < AA; ++a) { lg[a] = __expf(lg[a] - mx); s += lg[a]; }
            const float inv = 1.f / s;
            float* pR = accR + (b * NN + i) * AA;
            float* pQ = accQ + (b * NN + i) * AA;
            for (int a = 0; a < AA; ++a) {
                atomicAdd(&pR[a], nmv * lg[a] * inv);
                atomicAdd(&pQ[a], nmv * qv[a]);
            }
        }
    }
}

// ------------------------------------------------------------------ finalize
__global__ __launch_bounds__(256)
void k_final(const float* __restrict__ smg, const float* __restrict__ accR,
             const float* __restrict__ accQ, float* __restrict__ out) {
    const int t = threadIdx.x;                 // b*N + n  (256 total)
    const int b = t >> 5, n = t & 31;
    float cnt = 0.f;
    for (int r = 0; r < NN; ++r) cnt += smg[(b * NN + r) * NN + n];
    cnt = fmaxf(cnt, 1.f);
    const float ic = 1.f / cnt;
    float probs[AA], s = 0.f;
    for (int a = 0; a < AA; ++a) {
        float er = accR[t * AA + a] * ic;
        float eq = accQ[t * AA + a] * ic;
        float p = (1.f - 0.05f) * er + 0.05f / AA;
        probs[a] = p; s += p;
        out[BB * NN * AA + t * AA + a] = eq;       // ens_q
        out[2 * BB * NN * AA + t * AA + a] = er;   // ens_raw
    }
    s = fmaxf(s, 1e-8f);
    for (int a = 0; a < AA; ++a) out[t * AA + a] = probs[a] / s;   // probs
}

// -------------------------------------------------------------------- launch
extern "C" void kernel_launch(void* const* d_in, const int* in_sizes, int n_in,
                              void* d_out, int out_size, void* d_ws, size_t ws_size,
                              hipStream_t stream) {
    (void)in_sizes; (void)n_in; (void)out_size; (void)ws_size;
    const float* obs   = (const float*)d_in[0];
    const float* ef    = (const float*)d_in[1];
    const float* adj   = (const float*)d_in[2];
    const float* sm    = (const float*)d_in[3];
    const float* ln_g  = (const float*)d_in[4];
    const float* ln_b  = (const float*)d_in[5];
    const float* w1    = (const float*)d_in[6];
    const float* b1    = (const float*)d_in[7];
    const float* w2    = (const float*)d_in[8];
    const float* b2    = (const float*)d_in[9];
    const float* e_w   = (const float*)d_in[10];
    const float* e_b   = (const float*)d_in[11];
    const float* nr_w  = (const float*)d_in[12];
    const float* nr_b  = (const float*)d_in[13];
    const float* er_w  = (const float*)d_in[14];
    const float* er_b  = (const float*)d_in[15];
    const float* no_w  = (const float*)d_in[16];
    const float* no_b  = (const float*)d_in[17];
    const float* eo_w  = (const float*)d_in[18];
    const float* eo_b  = (const float*)d_in[19];
    const float* pol_w = (const float*)d_in[20];
    const float* pol_b = (const float*)d_in[21];
    const float* q_w   = (const float*)d_in[22];
    const float* q_b   = (const float*)d_in[23];

    float* ws    = (float*)d_ws;
    float* node0 = ws;                               // B*N*H
    float* edge0 = node0 + BB * NN * HH;             // B*N*N*E
    float* accR  = edge0 + BB * NN * NN * EE;        // B*N*A
    float* accQ  = accR + BB * NN * AA;              // B*N*A

    k_zero<<<(2 * BB * NN * AA + 255) / 256, 256, 0, stream>>>(accR, 2 * BB * NN * AA);
    k_enc_nodes<<<BB * NN, 128, 0, stream>>>(obs, ln_g, ln_b, w1, b1, w2, b2, node0);
    k_enc_edges<<<BB * NN, 256, 0, stream>>>(ef, e_w, e_b, edge0);
    k_gnn_root<<<BB * NN, 256, 0, stream>>>(adj, sm, nr_w, nr_b, er_w, er_b,
                                            no_w, no_b, eo_w, eo_b,
                                            pol_w, pol_b, q_w, q_b,
                                            node0, edge0, accR, accQ);
    k_final<<<1, 256, 0, stream>>>(sm, accR, accQ, (float*)d_out);
}
